// SimpleKANLayer_43920335569623
// MI455X (gfx1250) — compile-verified, matrix operand order checked by reading, not run on hardware
//
#include <hip/hip_runtime.h>

typedef __attribute__((ext_vector_type(16))) _Float16 v16h;
typedef __attribute__((ext_vector_type(8)))  float    v8f;
typedef __attribute__((ext_vector_type(4)))  int      v4i;

typedef __attribute__((address_space(1))) v4i* g4_ptr;   // global int4*
typedef __attribute__((address_space(3))) v4i* l4_ptr;   // LDS int4*

#define B_SZ   4096
#define IN_SZ  1024
#define OUT_SZ 1024
#define G_SZ   5
#define K_SZ   (IN_SZ * G_SZ)   // 5120

#define BM 128
#define BN 128
#define BK 32
#define BKP 40                    // 80B row stride: 16B-aligned, conflict-free
#define KSTEPS (K_SZ / BK)        // 160

#if defined(__has_builtin)
#if __has_builtin(__builtin_amdgcn_global_load_async_to_lds_b128) && \
    __has_builtin(__builtin_amdgcn_s_wait_asynccnt)
#define USE_ASYNC_LDS 1
#endif
#endif

// ---------------------------------------------------------------------------
// Prep 1: expand x -> W[B][K] (f16). Per (b,i): 5 entries, 2 nonzero.
// ---------------------------------------------------------------------------
__global__ __launch_bounds__(256) void kan_prep_w(const float* __restrict__ x,
                                                  _Float16* __restrict__ W) {
  int tid = blockIdx.x * blockDim.x + threadIdx.x;   // one per (b, i)
  if (tid >= B_SZ * IN_SZ) return;

  float xv = x[tid];
  float xc = fminf(fmaxf(xv, -1.0f), 1.0f);
  const float h = 2.0f / (float)(G_SZ - 1);          // 0.5
  int idx = (int)floorf((xc + 1.0f) / h);
  idx = idx < 0 ? 0 : (idx > G_SZ - 2 ? G_SZ - 2 : idx);
  float left  = -1.0f + (float)idx * h;
  float t     = (xc - left) / h;
  float valid = (xc < 1.0f) ? 1.0f : 0.0f;

  float w[G_SZ];
#pragma unroll
  for (int g = 0; g < G_SZ; ++g) w[g] = 0.0f;
  w[idx]     = (1.0f - t) * valid;
  w[idx + 1] = t * valid;

  _Float16* row = W + (size_t)tid * G_SZ;            // b*K + i*G
#pragma unroll
  for (int g = 0; g < G_SZ; ++g) row[g] = (_Float16)w[g];
}

// ---------------------------------------------------------------------------
// Prep 2: fold base_weight into coefficients -> M[OUT][K] (f16), row-major.
// ---------------------------------------------------------------------------
__global__ __launch_bounds__(256) void kan_prep_m(const float* __restrict__ coef,
                                                  const float* __restrict__ bw,
                                                  _Float16* __restrict__ M) {
  int tid = blockIdx.x * blockDim.x + threadIdx.x;   // one per (o, i)
  if (tid >= OUT_SZ * IN_SZ) return;
  float w = bw[tid];
  const float* c = coef + (size_t)tid * G_SZ;
  _Float16* row = M + (size_t)tid * G_SZ;            // o*K + i*G
#pragma unroll
  for (int g = 0; g < G_SZ; ++g) row[g] = (_Float16)(c[g] * w);
}

// ---------------------------------------------------------------------------
// Fragment assembly per CDNA5 16-bit A/B VGPR layout (wave32):
//   lane L<16 : row/col = L,     K in {0..7} u {16..23}
//   lane L>=16: row/col = L-16,  K in {8..15} u {24..31}
// ---------------------------------------------------------------------------
__device__ inline v16h load_frag(const _Float16* row_base, int khalf) {
  union { v16h v; uint4 q[2]; } u;
  u.q[0] = *(const uint4*)(row_base + khalf);
  u.q[1] = *(const uint4*)(row_base + khalf + 16);
  return u.v;
}

// ---------------------------------------------------------------------------
// GEMM: out[B][OUT] = W[B][K] * M[OUT][K]^T, f16 in / f32 accumulate (WMMA).
// 128x128 block tile, BK=32, 8 wave32 waves (2x4), 64x32 per wave.
// Double-buffered LDS fed by async global->LDS when available.
// ---------------------------------------------------------------------------
__global__ __launch_bounds__(256) void kan_gemm(const _Float16* __restrict__ W,
                                                const _Float16* __restrict__ M,
                                                float* __restrict__ out) {
  __shared__ _Float16 As[2][BM][BKP];
  __shared__ _Float16 Bs[2][BN][BKP];

  const int tid  = threadIdx.x;
  const int wave = tid >> 5;         // wave32
  const int lane = tid & 31;
  const int wm   = wave >> 2;        // 0..1
  const int wn   = wave & 3;         // 0..3
  const int bm   = blockIdx.x * BM;
  const int bn   = blockIdx.y * BN;

  // staging: thread t -> row t/2, halfs (t&1)*16 .. +15  (two 16B chunks)
  const int lr = tid >> 1;
  const int lc = (tid & 1) * 16;

  const _Float16* gA = W + (size_t)(bm + lr) * K_SZ + lc;
  const _Float16* gB = M + (size_t)(bn + lr) * K_SZ + lc;

  v8f acc[4][2];
#pragma unroll
  for (int mt = 0; mt < 4; ++mt)
#pragma unroll
    for (int nt = 0; nt < 2; ++nt) acc[mt][nt] = (v8f)(0.0f);

  const int row16 = lane & 15;
  const int khalf = (lane < 16) ? 0 : 8;

  auto do_tile = [&](int buf) {
    v16h afrag[4], bfrag[2];
#pragma unroll
    for (int mt = 0; mt < 4; ++mt)
      afrag[mt] = load_frag(&As[buf][wm * 64 + mt * 16 + row16][0], khalf);
#pragma unroll
    for (int nt = 0; nt < 2; ++nt)
      bfrag[nt] = load_frag(&Bs[buf][wn * 32 + nt * 16 + row16][0], khalf);
#pragma unroll
    for (int mt = 0; mt < 4; ++mt)
#pragma unroll
      for (int nt = 0; nt < 2; ++nt)
        acc[mt][nt] = __builtin_amdgcn_wmma_f32_16x16x32_f16(
            false, afrag[mt], false, bfrag[nt],
            (short)0, acc[mt][nt], false, false);
  };

#if defined(USE_ASYNC_LDS)
  // ---- async global->LDS, double-buffered; 4 in-flight ops per tile ----
  auto issue = [&](int buf, int k0) {
    __builtin_amdgcn_global_load_async_to_lds_b128(
        (g4_ptr)(gA + k0),     (l4_ptr)&As[buf][lr][lc], 0, 0);
    __builtin_amdgcn_global_load_async_to_lds_b128(
        (g4_ptr)(gA + k0 + 8), (l4_ptr)&As[buf][lr][lc + 8], 0, 0);
    __builtin_amdgcn_global_load_async_to_lds_b128(
        (g4_ptr)(gB + k0),     (l4_ptr)&Bs[buf][lr][lc], 0, 0);
    __builtin_amdgcn_global_load_async_to_lds_b128(
        (g4_ptr)(gB + k0 + 8), (l4_ptr)&Bs[buf][lr][lc + 8], 0, 0);
  };

  issue(0, 0);
  for (int ks = 0; ks < KSTEPS - 1; ++ks) {
    issue((ks & 1) ^ 1, (ks + 1) * BK);                // prefetch next tile
    __builtin_amdgcn_s_wait_asynccnt(4);               // tile ks landed
    __syncthreads();
    do_tile(ks & 1);
    __syncthreads();                                   // reads done before reuse
  }
  __builtin_amdgcn_s_wait_asynccnt(0);
  __syncthreads();
  do_tile((KSTEPS - 1) & 1);
#else
  // ---- synchronous fallback (full 32B per thread per matrix) ----
  for (int ks = 0; ks < KSTEPS; ++ks) {
    const int k0 = ks * BK;
    uint4 a0 = *(const uint4*)(gA + k0);
    uint4 a1 = *(const uint4*)(gA + k0 + 8);
    uint4 b0 = *(const uint4*)(gB + k0);
    uint4 b1 = *(const uint4*)(gB + k0 + 8);
    __syncthreads();
    *(uint4*)&As[0][lr][lc]     = a0;
    *(uint4*)&As[0][lr][lc + 8] = a1;
    *(uint4*)&Bs[0][lr][lc]     = b0;
    *(uint4*)&Bs[0][lr][lc + 8] = b1;
    __syncthreads();
    do_tile(0);
  }
#endif

  // Writeback: C layout — VGPR r, lanes 0-15: M=r, lanes 16-31: M=r+8; N=lane&15
  const int mhi = (lane >> 4) * 8;
#pragma unroll
  for (int mt = 0; mt < 4; ++mt) {
#pragma unroll
    for (int nt = 0; nt < 2; ++nt) {
      const int row0 = bm + wm * 64 + mt * 16 + mhi;
      const int col  = bn + wn * 32 + nt * 16 + row16;
      v8f c = acc[mt][nt];
#pragma unroll
      for (int r = 0; r < 8; ++r)
        out[(size_t)(row0 + r) * OUT_SZ + col] = c[r];
    }
  }
}

// ---------------------------------------------------------------------------
extern "C" void kernel_launch(void* const* d_in, const int* in_sizes, int n_in,
                              void* d_out, int out_size, void* d_ws, size_t ws_size,
                              hipStream_t stream) {
  const float* x    = (const float*)d_in[0];   // [B, IN]
  const float* coef = (const float*)d_in[1];   // [OUT, IN, G]
  const float* bw   = (const float*)d_in[2];   // [OUT, IN]
  float* out        = (float*)d_out;           // [B, OUT]

  char* ws = (char*)d_ws;
  _Float16* W = (_Float16*)ws;                                            // 41.9 MB
  _Float16* M = (_Float16*)(ws + (size_t)B_SZ * K_SZ * sizeof(_Float16)); // 10.5 MB

  kan_prep_w<<<(B_SZ * IN_SZ) / 256, 256, 0, stream>>>(x, W);
  kan_prep_m<<<(OUT_SZ * IN_SZ) / 256, 256, 0, stream>>>(coef, bw, M);

  dim3 grid(B_SZ / BM, OUT_SZ / BN);   // 32 x 8 = 256 workgroups
  kan_gemm<<<grid, 256, 0, stream>>>(W, M, out);
}
